// DecoderPart2Transformer_38079180046745
// MI455X (gfx1250) — matrix operand order repaired and checked by reading, not compiled, not run on hardware
//
#include <hip/hip_runtime.h>
#include <hip/hip_bf16.h>
#include <math.h>

#define LAYERS 8
#define BBATCH 128
#define NTOK   64
#define PPAST  8
#define HDIM   1024
#define NHEADS 16
#define NKVH   8
#define HDH    64
#define FFDIM  3072
#define WWIN   72
#define MTOK   (BBATCH*NTOK)   /* 8192 tokens */

typedef __bf16 bf16_t;
typedef bf16_t v16bf __attribute__((ext_vector_type(16)));
typedef float  v8f   __attribute__((ext_vector_type(8)));

union FragU { uint4 u[2]; v16bf v; };

__device__ __forceinline__ unsigned short f2bf(float f){
    union { float f; unsigned u; } x; x.f = f;
    unsigned r = x.u + 0x7FFFu + ((x.u >> 16) & 1u);   // RTNE
    return (unsigned short)(r >> 16);
}
__device__ __forceinline__ unsigned pack2bf(float lo, float hi){
    return (unsigned)f2bf(lo) | ((unsigned)f2bf(hi) << 16);
}

__device__ __forceinline__ float wsum(float x){
#pragma unroll
    for (int o = 16; o > 0; o >>= 1) x += __shfl_xor(x, o, 32);
    return x;
}
__device__ __forceinline__ float wmaxr(float x){
#pragma unroll
    for (int o = 16; o > 0; o >>= 1) x = fmaxf(x, __shfl_xor(x, o, 32));
    return x;
}

// Async copy of 16B from global memory into LDS (ASYNCcnt-tracked DMA path).
// ldsOff = byte offset inside the workgroup LDS allocation (low 32 bits of
// the generic pointer to a __shared__ object).
__device__ __forceinline__ void async_copy_b128(unsigned ldsOff, const void* gaddr){
    asm volatile("global_load_async_to_lds_b128 %0, %1, off"
                 :: "v"(ldsOff), "v"(gaddr) : "memory");
}
__device__ __forceinline__ void wait_async0(void){
    asm volatile("s_wait_asynccnt 0x0" ::: "memory");
}
__device__ __forceinline__ unsigned lds_off(const void* p){
    return (unsigned)(size_t)p;        // low 32 bits of generic ptr == LDS addr
}

// ---------------------------------------------------------------------------
// bf16-WMMA GEMM: C[M,Nc](f32) = A[M,K](f32) @ Bw[K,Nc](f32)
// MODE 0: C = A@B   MODE 1: C += A@B   MODE 2: transposed [B,H,N] store
// Block tile 128x128x32, 256 threads / 8 waves; each wave owns a 32x64
// sub-tile -> 8 x v_wmma_f32_16x16x32_bf16 per K-step.
// ---------------------------------------------------------------------------
template <int MODE>
__global__ __launch_bounds__(256) void gemm_bf16_kernel(
        const float* __restrict__ A, const float* __restrict__ Bw,
        float* __restrict__ C, int M, int Nc, int K)
{
    __shared__ unsigned short As[128][40];   // [row][k], +8 pad, 16B aligned rows
    __shared__ unsigned short Bs[128][40];   // transposed: [col][k]

    const int tid  = threadIdx.x;
    const int rowBase = blockIdx.y * 128;
    const int colBase = blockIdx.x * 128;
    const int lane = tid & 31, wave = tid >> 5;
    const int half = lane >> 4, lrow = lane & 15;
    const int wm = wave >> 1, wn = wave & 1;       // 4x2 wave grid
    // A loader: rows lr+{0,32,64,96}, cols lc..lc+3 (pairs packed along K)
    const int lr = tid >> 3,  lc = (tid & 7) << 2;
    // B loader: K rows kq..kq+3, cols bc4..bc4+3 (pairs packed along K)
    const int kq = (tid >> 5) << 2, bc4 = (tid & 31) << 2;

    v8f acc[2][4];
#pragma unroll
    for (int i = 0; i < 2; ++i)
#pragma unroll
        for (int j = 0; j < 4; ++j) acc[i][j] = {};

    for (int k0 = 0; k0 < K; k0 += 32){
        float4 a0 = *(const float4*)(A + (size_t)(rowBase + lr      ) * K + k0 + lc);
        float4 a1 = *(const float4*)(A + (size_t)(rowBase + lr +  32) * K + k0 + lc);
        float4 a2 = *(const float4*)(A + (size_t)(rowBase + lr +  64) * K + k0 + lc);
        float4 a3 = *(const float4*)(A + (size_t)(rowBase + lr +  96) * K + k0 + lc);
        float4 b0 = *(const float4*)(Bw + (size_t)(k0 + kq    ) * Nc + colBase + bc4);
        float4 b1 = *(const float4*)(Bw + (size_t)(k0 + kq + 1) * Nc + colBase + bc4);
        float4 b2 = *(const float4*)(Bw + (size_t)(k0 + kq + 2) * Nc + colBase + bc4);
        float4 b3 = *(const float4*)(Bw + (size_t)(k0 + kq + 3) * Nc + colBase + bc4);

        if (k0 + 32 < K){   // pull next K-tile toward cache while WMMAs run
            __builtin_prefetch(A  + (size_t)(rowBase + lr) * K + k0 + 32 + lc, 0, 0);
            __builtin_prefetch(Bw + (size_t)(k0 + 32 + kq) * Nc + colBase + bc4, 0, 0);
        }

        *(unsigned*)&As[lr     ][lc  ] = pack2bf(a0.x, a0.y);
        *(unsigned*)&As[lr     ][lc+2] = pack2bf(a0.z, a0.w);
        *(unsigned*)&As[lr + 32][lc  ] = pack2bf(a1.x, a1.y);
        *(unsigned*)&As[lr + 32][lc+2] = pack2bf(a1.z, a1.w);
        *(unsigned*)&As[lr + 64][lc  ] = pack2bf(a2.x, a2.y);
        *(unsigned*)&As[lr + 64][lc+2] = pack2bf(a2.z, a2.w);
        *(unsigned*)&As[lr + 96][lc  ] = pack2bf(a3.x, a3.y);
        *(unsigned*)&As[lr + 96][lc+2] = pack2bf(a3.z, a3.w);

        // Bs[col][k]: pack (k,k+1) pairs -> dword stores
        *(unsigned*)&Bs[bc4 + 0][kq  ] = pack2bf(b0.x, b1.x);
        *(unsigned*)&Bs[bc4 + 0][kq+2] = pack2bf(b2.x, b3.x);
        *(unsigned*)&Bs[bc4 + 1][kq  ] = pack2bf(b0.y, b1.y);
        *(unsigned*)&Bs[bc4 + 1][kq+2] = pack2bf(b2.y, b3.y);
        *(unsigned*)&Bs[bc4 + 2][kq  ] = pack2bf(b0.z, b1.z);
        *(unsigned*)&Bs[bc4 + 2][kq+2] = pack2bf(b2.z, b3.z);
        *(unsigned*)&Bs[bc4 + 3][kq  ] = pack2bf(b0.w, b1.w);
        *(unsigned*)&Bs[bc4 + 3][kq+2] = pack2bf(b2.w, b3.w);
        __syncthreads();

        // B fragments: col = lrow, K = {half*8.., 16+half*8..} per ISA table
        FragU bf[4];
#pragma unroll
        for (int nj = 0; nj < 4; ++nj){
            const int cc = wn*64 + nj*16 + lrow;
            bf[nj].u[0] = *(const uint4*)&Bs[cc][half*8];
            bf[nj].u[1] = *(const uint4*)&Bs[cc][16 + half*8];
        }
#pragma unroll
        for (int mi = 0; mi < 2; ++mi){
            // A fragment: row = lrow, K = {half*8.., 16+half*8..} per ISA table
            FragU a;
            const int ar = wm*32 + mi*16 + lrow;
            a.u[0] = *(const uint4*)&As[ar][half*8];
            a.u[1] = *(const uint4*)&As[ar][16 + half*8];
#pragma unroll
            for (int nj = 0; nj < 4; ++nj)
                acc[mi][nj] = __builtin_amdgcn_wmma_f32_16x16x32_bf16(
                    false, a.v, false, bf[nj].v, (short)0, acc[mi][nj], false, false);
        }
        __syncthreads();
    }

#pragma unroll
    for (int mi = 0; mi < 2; ++mi){
        const int gm0 = rowBase + wm*32 + mi*16 + half*8;   // C VGPR r -> M = gm0+r
#pragma unroll
        for (int nj = 0; nj < 4; ++nj){
            const int gn = colBase + wn*64 + nj*16 + lrow;
            if (MODE == 2){
                // out[(b*Nc + gn)*NTOK + n], b = gm>>6, n = gm&63 (gm0%8==0)
                float* p = C + ((size_t)(gm0 >> 6) * Nc + gn) * NTOK + (gm0 & 63);
#pragma unroll
                for (int r = 0; r < 8; ++r) p[r] = acc[mi][nj][r];
            } else {
                float* p = C + (size_t)gm0 * Nc + gn;
#pragma unroll
                for (int r = 0; r < 8; ++r){
                    if (MODE == 1) p[(size_t)r * Nc] += acc[mi][nj][r];
                    else           p[(size_t)r * Nc]  = acc[mi][nj][r];
                }
            }
        }
    }
}

// ---------------------------------------------------------------------------
// RMS norm over H=1024, block per row.
// ---------------------------------------------------------------------------
__global__ __launch_bounds__(256) void rmsnorm_kernel(
        const float* __restrict__ in, const float* __restrict__ w,
        float* __restrict__ out)
{
    const int row = blockIdx.x, tid = threadIdx.x;
    const float* x = in + (size_t)row * HDIM;
    float ss = 0.f;
    for (int i = tid; i < HDIM; i += 256){ float v = x[i]; ss += v*v; }
    __shared__ float red[256];
    red[tid] = ss; __syncthreads();
    for (int s = 128; s > 0; s >>= 1){ if (tid < s) red[tid] += red[tid+s]; __syncthreads(); }
    const float r = rsqrtf(red[0] * (1.0f/HDIM) + 1e-6f);
    for (int i = tid; i < HDIM; i += 256) out[(size_t)row*HDIM + i] = x[i] * r * w[i];
}

// ---------------------------------------------------------------------------
// Head RMS (qn) + RoPE, in place, one wave per (token, q-head).
// lane j handles pair (d=j, d=j+32): rope pairs share freq f_j.
// ---------------------------------------------------------------------------
#define LN1E6_OVER_32 0.4317347049363836f

__global__ __launch_bounds__(256) void rope_q_kernel(
        float* __restrict__ q, const float* __restrict__ qn)
{
    const int gw   = (blockIdx.x * 256 + threadIdx.x) >> 5;
    const int lane = threadIdx.x & 31;
    const int h = gw % NHEADS, tok = gw / NHEADS;
    const int n = tok % NTOK;
    float* base = q + (size_t)tok * HDIM + h * HDH;
    const float x0 = base[lane], x1 = base[lane + 32];
    const float ss = wsum(x0*x0 + x1*x1);
    const float r  = rsqrtf(ss * (1.0f/HDH) + 1e-6f);
    const float y0 = x0 * r * qn[lane], y1 = x1 * r * qn[lane + 32];
    const float f  = (float)(PPAST + n) * expf(-(float)lane * LN1E6_OVER_32);
    const float c = cosf(f), s = sinf(f);
    base[lane]      = y0*c - y1*s;
    base[lane + 32] = y1*c + y0*s;
}

// Head RMS (kn) + RoPE for k, plus v passthrough, writing straight into the
// output KV caches at t = PPAST + n. One wave per (token, kv-head).
__global__ __launch_bounds__(256) void rope_kv_kernel(
        const float* __restrict__ kraw, const float* __restrict__ vraw,
        const float* __restrict__ kn,
        float* __restrict__ ksC, float* __restrict__ vsC)
{
    const int gw   = (blockIdx.x * 256 + threadIdx.x) >> 5;
    const int lane = threadIdx.x & 31;
    const int kv = gw % NKVH, tok = gw / NKVH;
    const int b = tok / NTOK, n = tok % NTOK;
    const float* kb = kraw + (size_t)tok * (NKVH*HDH) + kv * HDH;
    const float x0 = kb[lane], x1 = kb[lane + 32];
    const float ss = wsum(x0*x0 + x1*x1);
    const float r  = rsqrtf(ss * (1.0f/HDH) + 1e-6f);
    const float y0 = x0 * r * kn[lane], y1 = x1 * r * kn[lane + 32];
    const float f  = (float)(PPAST + n) * expf(-(float)lane * LN1E6_OVER_32);
    const float c = cosf(f), s = sinf(f);
    const size_t dst = ((size_t)(b*NKVH + kv) * WWIN + (PPAST + n)) * HDH;
    ksC[dst + lane]      = y0*c - y1*s;
    ksC[dst + lane + 32] = y1*c + y0*s;
    const float* vb = vraw + (size_t)tok * (NKVH*HDH) + kv * HDH;
    vsC[dst + lane]      = vb[lane];
    vsC[dst + lane + 32] = vb[lane + 32];
}

// Copy past_k/past_v [L,B,NKV,P,HD] into cache slots t<P for all layers.
__global__ void past_copy_kernel(const float* __restrict__ pk,
                                 const float* __restrict__ pv,
                                 float* __restrict__ ksC, float* __restrict__ vsC)
{
    const size_t total = (size_t)LAYERS*BBATCH*NKVH*PPAST*HDH;
    for (size_t i = blockIdx.x*(size_t)blockDim.x + threadIdx.x; i < total;
         i += (size_t)gridDim.x*blockDim.x){
        const int d = (int)(i & 63);
        const int t = (int)((i >> 6) & 7);
        const size_t g = i >> 9;                      // (l*B+b)*NKV+kv
        const size_t dst = (g*WWIN + t)*HDH + d;
        ksC[dst] = pk[i];
        vsC[dst] = pv[i];
    }
}

// g = silu(g) * u
__global__ void silu_mul_kernel(float* __restrict__ g, const float* __restrict__ u)
{
    const size_t total = (size_t)MTOK*FFDIM;
    for (size_t i = blockIdx.x*(size_t)blockDim.x + threadIdx.x; i < total;
         i += (size_t)gridDim.x*blockDim.x){
        const float x = g[i];
        g[i] = (x / (1.f + expf(-x))) * u[i];
    }
}

// ---------------------------------------------------------------------------
// GQA attention, one block per (b, kv-head). K/V/Q staged in padded LDS via
// the CDNA5 async global->LDS DMA path (ASYNCcnt), 8 query rows per wave,
// exact reference masking/softmax.
// o written directly in [B,N,NH*HD] layout for the wo GEMM.
// ---------------------------------------------------------------------------
__global__ __launch_bounds__(256) void attn_kernel(
        const float* __restrict__ q,
        const float* __restrict__ ksC, const float* __restrict__ vsC,
        float* __restrict__ o)
{
    __shared__ float ks[WWIN][HDH+4];     // +4 pad (16B) keeps chunks aligned
    __shared__ float vs[WWIN][HDH+4];
    __shared__ float qs[NTOK][HDH+4];
    __shared__ float pbuf[8][WWIN];

    const int b = blockIdx.x / NKVH, kv = blockIdx.x % NKVH;
    const int tid = threadIdx.x, lane = tid & 31, wave = tid >> 5;
    const size_t kvbase = (size_t)(b*NKVH + kv) * WWIN * HDH;

    // Async-stage K and V tiles: rows are 64 contiguous floats -> 16B chunks.
    for (int c = tid; c < (WWIN*HDH)/4; c += 256){
        const int row = c >> 4, off = (c & 15) << 2;          // 16 chunks/row
        async_copy_b128(lds_off(&ks[row][off]), ksC + kvbase + row*HDH + off);
        async_copy_b128(lds_off(&vs[row][off]), vsC + kvbase + row*HDH + off);
    }

    for (int gidx = 0; gidx < 2; ++gidx){
        const int h = kv*2 + gidx;
        // qs rows are 64 contiguous floats of q[token].
        for (int c = tid; c < (NTOK*HDH)/4; c += 256){
            const int row = c >> 4, off = (c & 15) << 2;
            async_copy_b128(lds_off(&qs[row][off]),
                            q + (size_t)(b*NTOK + row)*HDIM + h*HDH + off);
        }
        wait_async0();
        __syncthreads();

#pragma unroll 1
        for (int r = 0; r < 8; ++r){
            const int n = wave*8 + r;
            // lane covers keys t = lane, lane+32, and (lane<8) lane+64
            float d0 = 0.f, d1 = 0.f, d2 = 0.f;
            for (int d = 0; d < HDH; ++d){
                const float qv = qs[n][d];
                d0 += qv * ks[lane     ][d];
                d1 += qv * ks[lane + 32][d];
                if (lane < 8) d2 += qv * ks[lane + 64][d];
            }
            const float s0 = d0*0.125f + ((lane      <= n + PPAST) ? 0.f : -10000.f);
            const float s1 = d1*0.125f + ((lane + 32 <= n + PPAST) ? 0.f : -10000.f);
            const float s2 = (lane < 8)
                ? d2*0.125f + ((lane + 64 <= n + PPAST) ? 0.f : -10000.f) : -1e30f;
            const float m  = wmaxr(fmaxf(fmaxf(s0, s1), s2));
            const float e0 = expf(s0 - m), e1 = expf(s1 - m);
            const float e2 = (lane < 8) ? expf(s2 - m) : 0.f;
            const float inv = 1.f / wsum(e0 + e1 + e2);
            pbuf[wave][lane]      = e0 * inv;
            pbuf[wave][lane + 32] = e1 * inv;
            if (lane < 8) pbuf[wave][lane + 64] = e2 * inv;
            asm volatile("s_wait_dscnt 0" ::: "memory");   // same-wave LDS RAW

            float a0 = 0.f, a1 = 0.f;
            for (int t = 0; t < WWIN; ++t){
                const float p = pbuf[wave][t];
                a0 += p * vs[t][lane];
                a1 += p * vs[t][lane + 32];
            }
            const size_t ob = (size_t)(b*NTOK + n)*HDIM + h*HDH;
            o[ob + lane]      = a0;
            o[ob + lane + 32] = a1;
        }
        __syncthreads();   // qs readers done before next head's async stores
    }
}

// ---------------------------------------------------------------------------
static inline void gemm_launch(const float* A, const float* B, float* C,
                               int M, int Nc, int K, int mode, hipStream_t s)
{
    dim3 grid(Nc/128, M/128), blk(256);
    if      (mode == 0) gemm_bf16_kernel<0><<<grid, blk, 0, s>>>(A, B, C, M, Nc, K);
    else if (mode == 1) gemm_bf16_kernel<1><<<grid, blk, 0, s>>>(A, B, C, M, Nc, K);
    else                gemm_bf16_kernel<2><<<grid, blk, 0, s>>>(A, B, C, M, Nc, K);
}

extern "C" void kernel_launch(void* const* d_in, const int* in_sizes, int n_in,
                              void* d_out, int out_size, void* d_ws, size_t ws_size,
                              hipStream_t stream)
{
    (void)in_sizes; (void)n_in; (void)out_size; (void)ws_size;
    const float* hidden = (const float*)d_in[0];
    const float* past_k = (const float*)d_in[1];
    const float* past_v = (const float*)d_in[2];
    const float* w_in   = (const float*)d_in[3];
    const float* ln1    = (const float*)d_in[4];
    const float* wq     = (const float*)d_in[5];
    const float* wk     = (const float*)d_in[6];
    const float* wv     = (const float*)d_in[7];
    const float* qn     = (const float*)d_in[8];
    const float* kn     = (const float*)d_in[9];
    const float* wo     = (const float*)d_in[10];
    const float* ln2    = (const float*)d_in[11];
    const float* wg     = (const float*)d_in[12];
    const float* wu     = (const float*)d_in[13];
    const float* wd     = (const float*)d_in[14];
    const float* norm_w = (const float*)d_in[15];
    const float* w_out  = (const float*)d_in[16];

    float* outH = (float*)d_out;                                  // [B,H,N]
    float* outK = outH + (size_t)BBATCH*HDIM*NTOK;                // [L,B,NKV,W,HD]
    float* outV = outK + (size_t)LAYERS*BBATCH*NKVH*WWIN*HDH;

    float* h   = (float*)d_ws;
    float* hn  = h   + (size_t)MTOK*HDIM;
    float* q   = hn  + (size_t)MTOK*HDIM;
    float* kvo = q   + (size_t)MTOK*HDIM;          // k|v raw, later o (aliased)
    float* kr  = kvo;
    float* vr  = kvo + (size_t)MTOK*(NKVH*HDH);
    float* o   = kvo;
    float* g   = kvo + (size_t)MTOK*HDIM;
    float* u   = g   + (size_t)MTOK*FFDIM;

    const dim3 blk(256);
    const size_t ksLayer = (size_t)BBATCH*NKVH*WWIN*HDH;

    past_copy_kernel<<<2048, blk, 0, stream>>>(past_k, past_v, outK, outV);
    gemm_launch(hidden, w_in, h, MTOK, HDIM, HDIM, 0, stream);

    for (int l = 0; l < LAYERS; ++l){
        rmsnorm_kernel<<<MTOK, blk, 0, stream>>>(h, ln1 + (size_t)l*HDIM, hn);
        gemm_launch(hn, wq + (size_t)l*HDIM*(NHEADS*HDH), q, MTOK, NHEADS*HDH, HDIM, 0, stream);
        gemm_launch(hn, wk + (size_t)l*HDIM*(NKVH*HDH), kr, MTOK, NKVH*HDH, HDIM, 0, stream);
        gemm_launch(hn, wv + (size_t)l*HDIM*(NKVH*HDH), vr, MTOK, NKVH*HDH, HDIM, 0, stream);
        rope_q_kernel <<<(MTOK*NHEADS)/8, blk, 0, stream>>>(q, qn + (size_t)l*HDH);
        rope_kv_kernel<<<(MTOK*NKVH )/8, blk, 0, stream>>>(
            kr, vr, kn + (size_t)l*HDH, outK + l*ksLayer, outV + l*ksLayer);
        attn_kernel<<<BBATCH*NKVH, blk, 0, stream>>>(
            q, outK + l*ksLayer, outV + l*ksLayer, o);
        gemm_launch(o, wo + (size_t)l*(NHEADS*HDH)*HDIM, h, MTOK, HDIM, NHEADS*HDH, 1, stream);

        rmsnorm_kernel<<<MTOK, blk, 0, stream>>>(h, ln2 + (size_t)l*HDIM, hn);
        gemm_launch(hn, wg + (size_t)l*HDIM*FFDIM, g, MTOK, FFDIM, HDIM, 0, stream);
        gemm_launch(hn, wu + (size_t)l*HDIM*FFDIM, u, MTOK, FFDIM, HDIM, 0, stream);
        silu_mul_kernel<<<4096, blk, 0, stream>>>(g, u);
        gemm_launch(g, wd + (size_t)l*FFDIM*HDIM, h, MTOK, HDIM, FFDIM, 1, stream);
    }

    rmsnorm_kernel<<<MTOK, blk, 0, stream>>>(h, norm_w, hn);
    gemm_launch(hn, w_out, outH, MTOK, HDIM, HDIM, 2, stream);   // [B,H,N] store
}
